// EncoderGroupATT_68375879352508
// MI455X (gfx1250) — matrix operand order repaired and checked by reading, not compiled, not run on hardware
//
#include <hip/hip_runtime.h>
#include <math.h>

typedef __attribute__((ext_vector_type(2)))  float  v2f;
typedef __attribute__((ext_vector_type(8)))  float  v8f;
typedef __attribute__((ext_vector_type(16))) __bf16 v16bf;

#define LN_EPS   1e-6f
#define NEGBIG   -1e9f
#define GF_RELU  1
#define GF_RES   2
#define GF_SPLIT 4

// ---------------- fp32 WMMA (16x16x4): attention ---------------------------
__device__ __forceinline__ v8f wmma4(v2f a, v2f b, v8f c) {
  return __builtin_amdgcn_wmma_f32_16x16x4_f32(false, a, false, b, (short)0, c,
                                               false, false);
}
__device__ __forceinline__ v2f ldA(const float* sm, int ld, int m0, int k0, int lane) {
  int m  = m0 + (lane & 15);
  int kb = k0 + ((lane >> 4) << 1);
  v2f r; r.x = sm[m * ld + kb]; r.y = sm[m * ld + kb + 1]; return r;
}
__device__ __forceinline__ v2f ldB(const float* sm, int ld, int k0, int n0, int lane) {
  int n  = n0 + (lane & 15);
  int kb = k0 + ((lane >> 4) << 1);
  v2f r; r.x = sm[kb * ld + n]; r.y = sm[(kb + 1) * ld + n]; return r;
}
__device__ __forceinline__ v2f ldBT(const float* sm, int ld, int k0, int n0, int lane) {
  int n  = n0 + (lane & 15);
  int kb = k0 + ((lane >> 4) << 1);
  v2f r; r.x = sm[n * ld + kb]; r.y = sm[n * ld + kb + 1]; return r;
}

// ---------------- bf16 WMMA (16x16x32): split-precision GEMMs ---------------
__device__ __forceinline__ v8f wmma_bf(v16bf a, v16bf b, v8f c) {
  return __builtin_amdgcn_wmma_f32_16x16x32_bf16(false, a, false, b, (short)0, c,
                                                 false, false);
}
union FragU { v16bf v; unsigned u[8]; };

// A fragment (16x32 bf16) from LDS row-major [64][34] bf16 (17 dwords/row).
__device__ __forceinline__ v16bf ldAbf(const unsigned* sm32, int m0, int lane) {
  const unsigned* rowp = sm32 + (m0 + (lane & 15)) * 17;
  const int kh = (lane >> 4) << 2;
  FragU f;
#pragma unroll
  for (int v = 0; v < 4; ++v) f.u[v]     = rowp[kh + v];
#pragma unroll
  for (int v = 0; v < 4; ++v) f.u[4 + v] = rowp[8 + kh + v];
  return f.v;
}
// B fragment (32x16 bf16) from LDS stored [N=64][34] bf16 (K-contiguous rows).
__device__ __forceinline__ v16bf ldBbf(const unsigned* sm32, int n0, int lane) {
  const unsigned* rowp = sm32 + (n0 + (lane & 15)) * 17;
  const int kh = (lane >> 4) << 2;
  FragU f;
#pragma unroll
  for (int v = 0; v < 4; ++v) f.u[v]     = rowp[kh + v];
#pragma unroll
  for (int v = 0; v < 4; ++v) f.u[4 + v] = rowp[8 + kh + v];
  return f.v;
}
__device__ __forceinline__ void split_bf(float x, __bf16* h, __bf16* l) {
  __bf16 hb = (__bf16)x;
  *h = hb;
  *l = (__bf16)(x - (float)hb);
}

// ---------------------------------------------------------------------------
// Weight pre-split + transpose: wh/wl[n*K + k] = hi/lo(W[k*Nn + n]).
// Run once per launch per tensor; GEMM staging then becomes pure dword copies.
// ---------------------------------------------------------------------------
__global__ __launch_bounds__(256) void wsplit_kernel(
    const float* __restrict__ W, __bf16* __restrict__ wh,
    __bf16* __restrict__ wl, int K, int Nn)
{
  size_t idx = (size_t)blockIdx.x * 256 + threadIdx.x;
  size_t total = (size_t)K * Nn;
  if (idx >= total) return;
  int n = (int)(idx / K);
  int k = (int)(idx % K);
  __bf16 h, l;
  split_bf(W[(size_t)k * Nn + n], &h, &l);
  wh[idx] = h; wl[idx] = l;
}

// ---------------------------------------------------------------------------
// LayerNorm variants. ddof=1 variance; y = a*(x-m)/(sqrt(var)+eps)+b.
// ---------------------------------------------------------------------------
__device__ __forceinline__ void ln_core(const float* xr, int tid, float* red,
                                        float v[4], float* meanp, float* invp)
{
  float s = 0.f;
#pragma unroll
  for (int j = 0; j < 4; ++j) { v[j] = xr[tid + j * 256]; s += v[j]; }
  red[tid] = s; __syncthreads();
  for (int off = 128; off > 0; off >>= 1) {
    if (tid < off) red[tid] += red[tid + off];
    __syncthreads();
  }
  float mean = red[0] * (1.f / 1024.f);
  __syncthreads();
  float s2 = 0.f;
#pragma unroll
  for (int j = 0; j < 4; ++j) { float d = v[j] - mean; s2 += d * d; }
  red[tid] = s2; __syncthreads();
  for (int off = 128; off > 0; off >>= 1) {
    if (tid < off) red[tid] += red[tid + off];
    __syncthreads();
  }
  float var = red[0] * (1.f / 1023.f);
  *meanp = mean;
  *invp = 1.f / (sqrtf(var) + LN_EPS);
}

__global__ __launch_bounds__(256) void ln_kernel(
    const float* __restrict__ x, const float* __restrict__ ga,
    const float* __restrict__ gb, float* __restrict__ y)
{
  __shared__ float red[256];
  const int tid = threadIdx.x;
  const size_t row = blockIdx.x;
  float v[4], mean, inv;
  ln_core(x + row * 1024, tid, red, v, &mean, &inv);
#pragma unroll
  for (int j = 0; j < 4; ++j) {
    int c = tid + j * 256;
    y[row * 1024 + c] = ga[c] * (v[j] - mean) * inv + gb[c];
  }
}

// LN emitting split bf16 hi/lo (GEMM-ready activations, no fp32 round-trip)
__global__ __launch_bounds__(256) void ln_split_kernel(
    const float* __restrict__ x, const float* __restrict__ ga,
    const float* __restrict__ gb, __bf16* __restrict__ oh,
    __bf16* __restrict__ ol)
{
  __shared__ float red[256];
  const int tid = threadIdx.x;
  const size_t row = blockIdx.x;
  float v[4], mean, inv;
  ln_core(x + row * 1024, tid, red, v, &mean, &inv);
#pragma unroll
  for (int j = 0; j < 4; ++j) {
    int c = tid + j * 256;
    float val = ga[c] * (v[j] - mean) * inv + gb[c];
    __bf16 h, l; split_bf(val, &h, &l);
    oh[row * 1024 + c] = h;
    ol[row * 1024 + c] = l;
  }
}

// ---------------------------------------------------------------------------
// Split-precision GEMM, bf16 hi/lo inputs (pre-split):
//   acc = Ah*Wh + Ah*Wl + Al*Wh  (fp32 accumulate)
// A: [M][K] bf16 row-major (hi/lo). W: [Nn][K] bf16 (pre-transposed hi/lo).
// Epilogue: +bias, optional relu, then fp32(+res) store OR split-bf16 store.
// ---------------------------------------------------------------------------
__global__ __launch_bounds__(256) void gemm_bf_kernel(
    const __bf16* __restrict__ Ahg, const __bf16* __restrict__ Alg,
    const __bf16* __restrict__ Whg, const __bf16* __restrict__ Wlg,
    const float* __restrict__ bias, const float* __restrict__ res,
    float* __restrict__ outf, __bf16* __restrict__ oh, __bf16* __restrict__ ol,
    int K, int Nn, int flags)
{
  __shared__ unsigned sAh[64 * 17];
  __shared__ unsigned sAl[64 * 17];
  __shared__ unsigned sWh[64 * 17];
  __shared__ unsigned sWl[64 * 17];
  const unsigned* Ah32 = (const unsigned*)Ahg;
  const unsigned* Al32 = (const unsigned*)Alg;
  const unsigned* Wh32 = (const unsigned*)Whg;
  const unsigned* Wl32 = (const unsigned*)Wlg;

  const int tid  = threadIdx.x;
  const int lane = tid & 31;
  const int wv   = tid >> 5;
  const int wrow = wv >> 1;
  const int wcol = wv & 1;
  const int row0 = blockIdx.y * 64;
  const int col0 = blockIdx.x * 64;

  v8f c0 = {}; v8f c1 = {};

  for (int k0 = 0; k0 < K; k0 += 32) {
#pragma unroll
    for (int it = 0; it < 4; ++it) {
      int id = tid + it * 256;                  // 0..1023 dwords per array
      int r = id >> 4;
      int j = id & 15;
      size_t gA = (((size_t)(row0 + r) * K + k0) >> 1) + j;
      size_t gW = (((size_t)(col0 + r) * K + k0) >> 1) + j;
      sAh[r * 17 + j] = Ah32[gA];
      sAl[r * 17 + j] = Al32[gA];
      sWh[r * 17 + j] = Wh32[gW];
      sWl[r * 17 + j] = Wl32[gW];
    }
    __syncthreads();

    v16bf ah  = ldAbf(sAh, wrow * 16, lane);
    v16bf al  = ldAbf(sAl, wrow * 16, lane);
    v16bf bh0 = ldBbf(sWh, wcol * 32,      lane);
    v16bf bl0 = ldBbf(sWl, wcol * 32,      lane);
    v16bf bh1 = ldBbf(sWh, wcol * 32 + 16, lane);
    v16bf bl1 = ldBbf(sWl, wcol * 32 + 16, lane);
    c0 = wmma_bf(ah, bh0, c0);
    c0 = wmma_bf(ah, bl0, c0);
    c0 = wmma_bf(al, bh0, c0);
    c1 = wmma_bf(ah, bh1, c1);
    c1 = wmma_bf(ah, bl1, c1);
    c1 = wmma_bf(al, bh1, c1);
    __syncthreads();
  }

  const int colA = col0 + wcol * 32 + (lane & 15);
  const int colB = colA + 16;
  const int rsel = (lane < 16) ? 0 : 8;
  const float bA = bias[colA];
  const float bB = bias[colB];
#pragma unroll
  for (int r = 0; r < 8; ++r) {
    int row = row0 + wrow * 16 + r + rsel;
    float v0 = c0[r] + bA;
    float v1 = c1[r] + bB;
    if (flags & GF_RELU) { v0 = fmaxf(v0, 0.f); v1 = fmaxf(v1, 0.f); }
    size_t iA = (size_t)row * Nn + colA;
    size_t iB = (size_t)row * Nn + colB;
    if (flags & GF_SPLIT) {
      __bf16 h, l;
      split_bf(v0, &h, &l); oh[iA] = h; ol[iA] = l;
      split_bf(v1, &h, &l); oh[iB] = h; ol[iB] = l;
    } else {
      if (flags & GF_RES) { v0 += res[iA]; v1 += res[iB]; }
      outf[iA] = v0;
      outf[iB] = v1;
    }
  }
}

// ---------------------------------------------------------------------------
// Flash attention (exact fp32 WMMA): one block per (b, h, 64-query tile).
// Output written as split bf16 hi/lo (feeds the o-proj GEMM directly).
// ---------------------------------------------------------------------------
__global__ __launch_bounds__(256) void attn_kernel(
    const float* __restrict__ Qg, const float* __restrict__ Kg,
    const float* __restrict__ Vg, const int* __restrict__ mask,
    __bf16* __restrict__ Oh, __bf16* __restrict__ Ol)
{
  const int Sq = 512, D = 1024;
  __shared__ float Qs[64 * 129];
  __shared__ float Ks[64 * 129];
  __shared__ float Vs[64 * 129];
  __shared__ float Ss[64 * 65];
  __shared__ float mrow[64], lrow[64], arow[64];

  const int tid  = threadIdx.x;
  const int lane = tid & 31;
  const int wv   = tid >> 5;
  const int wrow = wv >> 1;
  const int wcol = wv & 1;
  const int q0 = blockIdx.x * 64;
  const int h  = blockIdx.y;
  const int b  = blockIdx.z;
  const int g  = h & 3;
  const float scale = 0.08838834764831845f;   // 1/sqrt(128)

  for (int idx = tid; idx < 64 * 128; idx += 256) {
    int r = idx >> 7, c = idx & 127;
    Qs[r * 129 + c] = Qg[(size_t)(b * Sq + q0 + r) * D + h * 128 + c];
  }
  if (tid < 64) { mrow[tid] = -3.0e38f; lrow[tid] = 0.f; }

  v8f o0 = {}, o1 = {}, o2 = {}, o3 = {};

  for (int kt = 0; kt < 8; ++kt) {
    __syncthreads();
    const int j0 = kt * 64;
    for (int idx = tid; idx < 64 * 128; idx += 256) {
      int r = idx >> 7, c = idx & 127;
      Ks[r * 129 + c] = Kg[(size_t)(b * Sq + j0 + r) * D + h * 128 + c];
      Vs[r * 129 + c] = Vg[(size_t)(b * Sq + j0 + r) * D + h * 128 + c];
    }
    __syncthreads();

    v8f s0 = {}, s1 = {};
#pragma unroll
    for (int kk = 0; kk < 128; kk += 4) {
      v2f a  = ldA(Qs, 129, wrow * 16, kk, lane);
      v2f b0 = ldBT(Ks, 129, kk, wcol * 32, lane);
      v2f b1 = ldBT(Ks, 129, kk, wcol * 32 + 16, lane);
      s0 = wmma4(a, b0, s0);
      s1 = wmma4(a, b1, s1);
    }
    {
      const int rsel = (lane < 16) ? 0 : 8;
      const int cA = wcol * 32 + (lane & 15);
#pragma unroll
      for (int r = 0; r < 8; ++r) {
        int rl = wrow * 16 + r + rsel;
        size_t mb = (size_t)((b * 4 + g) * Sq + (q0 + rl)) * Sq + j0;
        float vA = s0[r] * scale, vB = s1[r] * scale;
        if (mask[mb + cA] == 0)      vA = NEGBIG;
        if (mask[mb + cA + 16] == 0) vB = NEGBIG;
        Ss[rl * 65 + cA]      = vA;
        Ss[rl * 65 + cA + 16] = vB;
      }
    }
    __syncthreads();

    if (tid < 64) {
      float mold = mrow[tid];
      float tmax = -3.0e38f;
      for (int j = 0; j < 64; ++j) tmax = fmaxf(tmax, Ss[tid * 65 + j]);
      float mnew = fmaxf(mold, tmax);
      float alpha = __expf(mold - mnew);
      float ssum = 0.f;
      for (int j = 0; j < 64; ++j) {
        float p = __expf(Ss[tid * 65 + j] - mnew);
        Ss[tid * 65 + j] = p;
        ssum += p;
      }
      mrow[tid] = mnew;
      lrow[tid] = lrow[tid] * alpha + ssum;
      arow[tid] = alpha;
    }
    __syncthreads();

    {
      const int rsel = (lane < 16) ? 0 : 8;
#pragma unroll
      for (int r = 0; r < 8; ++r) {
        float al = arow[wrow * 16 + r + rsel];
        o0[r] *= al; o1[r] *= al; o2[r] *= al; o3[r] *= al;
      }
    }
#pragma unroll
    for (int kk = 0; kk < 64; kk += 4) {
      v2f a  = ldA(Ss, 65, wrow * 16, kk, lane);
      v2f b0 = ldB(Vs, 129, kk, wcol * 64, lane);
      v2f b1 = ldB(Vs, 129, kk, wcol * 64 + 16, lane);
      v2f b2 = ldB(Vs, 129, kk, wcol * 64 + 32, lane);
      v2f b3 = ldB(Vs, 129, kk, wcol * 64 + 48, lane);
      o0 = wmma4(a, b0, o0);
      o1 = wmma4(a, b1, o1);
      o2 = wmma4(a, b2, o2);
      o3 = wmma4(a, b3, o3);
    }
  }
  __syncthreads();

  const int rsel = (lane < 16) ? 0 : 8;
#pragma unroll
  for (int r = 0; r < 8; ++r) {
    int rl = wrow * 16 + r + rsel;
    float inv = 1.f / lrow[rl];
    size_t ob = (size_t)(b * Sq + q0 + rl) * D + h * 128 + wcol * 64 + (lane & 15);
    __bf16 hh, ll;
    split_bf(o0[r] * inv, &hh, &ll); Oh[ob + 0]  = hh; Ol[ob + 0]  = ll;
    split_bf(o1[r] * inv, &hh, &ll); Oh[ob + 16] = hh; Ol[ob + 16] = ll;
    split_bf(o2[r] * inv, &hh, &ll); Oh[ob + 32] = hh; Ol[ob + 32] = ll;
    split_bf(o3[r] * inv, &hh, &ll); Oh[ob + 48] = hh; Ol[ob + 48] = ll;
  }
}

// ---------------------------------------------------------------------------
extern "C" void kernel_launch(void* const* d_in, const int* in_sizes, int n_in,
                              void* d_out, int out_size, void* d_ws, size_t ws_size,
                              hipStream_t stream) {
  const int B = 16, S = 512, D = 1024, F = 2048, NL = 2;
  const int M = B * S;
  const float* x    = (const float*)d_in[0];
  const int*   mask = (const int*)  d_in[1];
  const float* Wq   = (const float*)d_in[2];
  const float* bq   = (const float*)d_in[3];
  const float* Wk   = (const float*)d_in[4];
  const float* bk   = (const float*)d_in[5];
  const float* Wv   = (const float*)d_in[6];
  const float* bv   = (const float*)d_in[7];
  const float* W1   = (const float*)d_in[8];
  const float* b1   = (const float*)d_in[9];
  const float* W2   = (const float*)d_in[10];
  const float* b2   = (const float*)d_in[11];
  const float* ln1a = (const float*)d_in[12];
  const float* ln1b = (const float*)d_in[13];
  const float* ln2a = (const float*)d_in[14];
  const float* ln2b = (const float*)d_in[15];
  const float* lnfa = (const float*)d_in[16];
  const float* lnfb = (const float*)d_in[17];
  float* out = (float*)d_out;
  float* ws  = (float*)d_ws;

  const size_t U = (size_t)M * D;               // fp32 elems per slab
  float* xbuf = ws;                             // residual stream (fp32)
  float* qb   = ws + U;                         // q fp32 (later: FFN-hidden hi)
  float* kb   = ws + 2 * U;                     // k fp32 (later: FFN-hidden lo)
  float* vb   = ws + 3 * U;                     // v fp32
  __bf16* Ah  = (__bf16*)(ws + 4 * U);          // activation hi (M x D)
  __bf16* Al  = Ah + U;                         // activation lo
  __bf16* wp  = Al + U;                         // pre-split weight region

  // ---- pre-split + transpose all weights (bf16 hi/lo, [Nn][K]) ----
  __bf16 *wqh[2], *wql[2], *wkh[2], *wkl[2], *wvh[2], *wvl[2];
  __bf16 *w1h[2], *w1l[2], *w2h[2], *w2l[2];
  for (int i = 0; i < NL; ++i) {
    const size_t DD = (size_t)D * D, DF = (size_t)D * F;
    wqh[i] = wp; wp += DD; wql[i] = wp; wp += DD;
    wkh[i] = wp; wp += DD; wkl[i] = wp; wp += DD;
    wvh[i] = wp; wp += DD; wvl[i] = wp; wp += DD;
    w1h[i] = wp; wp += DF; w1l[i] = wp; wp += DF;
    w2h[i] = wp; wp += DF; w2l[i] = wp; wp += DF;
    wsplit_kernel<<<(int)(DD / 256), 256, 0, stream>>>(Wq + i * DD, wqh[i], wql[i], D, D);
    wsplit_kernel<<<(int)(DD / 256), 256, 0, stream>>>(Wk + i * DD, wkh[i], wkl[i], D, D);
    wsplit_kernel<<<(int)(DD / 256), 256, 0, stream>>>(Wv + i * DD, wvh[i], wvl[i], D, D);
    wsplit_kernel<<<(int)(DF / 256), 256, 0, stream>>>(W1 + i * DF, w1h[i], w1l[i], D, F);
    wsplit_kernel<<<(int)(DF / 256), 256, 0, stream>>>(W2 + i * DF, w2h[i], w2l[i], F, D);
  }

  hipMemcpyAsync(xbuf, x, U * sizeof(float), hipMemcpyDeviceToDevice, stream);

  dim3 blk(256);
  dim3 gDD(D / 64, M / 64);
  dim3 gDF(F / 64, M / 64);
  dim3 gFD(D / 64, M / 64);
  dim3 gAT(S / 64, 8, B);
  __bf16* Hh = (__bf16*)qb;                     // FFN hidden hi (M x F) aliases q
  __bf16* Hl = (__bf16*)kb;                     // FFN hidden lo aliases k

  for (int i = 0; i < NL; ++i) {
    ln_split_kernel<<<M, blk, 0, stream>>>(xbuf, ln1a + i * D, ln1b + i * D, Ah, Al);
    gemm_bf_kernel<<<gDD, blk, 0, stream>>>(Ah, Al, wqh[i], wql[i], bq + i * D,
                                            nullptr, qb, nullptr, nullptr, D, D, 0);
    gemm_bf_kernel<<<gDD, blk, 0, stream>>>(Ah, Al, wkh[i], wkl[i], bk + i * D,
                                            nullptr, kb, nullptr, nullptr, D, D, 0);
    gemm_bf_kernel<<<gDD, blk, 0, stream>>>(Ah, Al, wvh[i], wvl[i], bv + i * D,
                                            nullptr, vb, nullptr, nullptr, D, D, 0);
    attn_kernel<<<gAT, blk, 0, stream>>>(qb, kb, vb, mask, Ah, Al);
    // output projection reuses the value linear (reference quirk), + residual
    gemm_bf_kernel<<<gDD, blk, 0, stream>>>(Ah, Al, wvh[i], wvl[i], bv + i * D,
                                            xbuf, xbuf, nullptr, nullptr, D, D, GF_RES);
    ln_split_kernel<<<M, blk, 0, stream>>>(xbuf, ln2a + i * D, ln2b + i * D, Ah, Al);
    gemm_bf_kernel<<<gDF, blk, 0, stream>>>(Ah, Al, w1h[i], w1l[i], b1 + i * F,
                                            nullptr, nullptr, Hh, Hl, D, F,
                                            GF_RELU | GF_SPLIT);
    gemm_bf_kernel<<<gFD, blk, 0, stream>>>(Hh, Hl, w2h[i], w2l[i], b2 + i * D,
                                            xbuf, xbuf, nullptr, nullptr, F, D, GF_RES);
  }
  ln_kernel<<<M, blk, 0, stream>>>(xbuf, lnfa, lnfb, out);
}
